// InterpretableMultiHeadAttention_66958540144666
// MI455X (gfx1250) — matrix operand order repaired and checked by reading, not compiled, not run on hardware
//
#include <hip/hip_runtime.h>
#include <hip/hip_bf16.h>

// ---------------- Types for CDNA5 WMMA ----------------
typedef __attribute__((ext_vector_type(16))) __bf16         v16bf;
typedef __attribute__((ext_vector_type(8)))  float          v8f;
typedef __attribute__((ext_vector_type(16))) unsigned short v16us;

#define BB 4
#define TT 2048
#define DD 1024
#define HH 16
#define DH 64

// ---------------- bf16 helpers ----------------
__device__ __forceinline__ unsigned short f2bf(float x) {
    unsigned u = __float_as_uint(x);
    unsigned r = u + 0x7FFFu + ((u >> 16) & 1u);   // round-to-nearest-even
    return (unsigned short)(r >> 16);
}
__device__ __forceinline__ float bf2f(unsigned short h) {
    return __uint_as_float(((unsigned)h) << 16);
}

union Frag {
    v16us us;
    uint4 q[2];
};

// A-matrix 16x32 bf16 fragment (row-major source, ld in elements).
// Order used here (B matches): lane m=lane&15, half=lane>>4; elems e -> K = half*16+e (contiguous).
__device__ __forceinline__ v16us load_a16x32(const unsigned short* base, int ld, int lane) {
    const int m = lane & 15, half = lane >> 4;
    const unsigned short* r = base + (size_t)m * ld + half * 16;
    Frag f;
    f.q[0] = *(const uint4*)(r);
    f.q[1] = *(const uint4*)(r + 8);
    return f.us;
}

// B-matrix 32x16 bf16 fragment where B[k,n] = SRC[n, k] (SRC row-major, ld elems).
// Order used here (A matches): lane n=lane&15, khalf=lane>>4; elems e -> K = khalf*16+e (contiguous).
__device__ __forceinline__ v16us load_b32x16(const unsigned short* base, int ld, int lane) {
    const int n = lane & 15, kh = lane >> 4;
    const unsigned short* r = base + (size_t)n * ld + kh * 16;
    Frag f;
    f.q[0] = ((const uint4*)r)[0];
    f.q[1] = ((const uint4*)r)[1];
    return f.us;
}

__device__ __forceinline__ v8f wmma_bf16(v16us a, v16us b, v8f c) {
    return __builtin_amdgcn_wmma_f32_16x16x32_bf16(
        false, __builtin_bit_cast(v16bf, a),
        false, __builtin_bit_cast(v16bf, b),
        (short)0, c, false, false);
}

// ---------------- fp32 -> bf16 convert ----------------
__global__ void cvt_bf16_kernel(const float* __restrict__ in, unsigned short* __restrict__ out, int n) {
    int i = blockIdx.x * blockDim.x + threadIdx.x;
    const int stride = gridDim.x * blockDim.x;
    for (; i < n; i += stride) out[i] = f2bf(in[i]);
}

// ---------------- GEMM: C[M,N] = A[M,K] * W[N,K]^T + bias, bf16 out ----------------
// vt=1: store transposed per batch of TT rows: C[b][n][t] (for V projection).
__global__ __launch_bounds__(256) void gemm_bf16out_kernel(
    const unsigned short* __restrict__ A, const unsigned short* __restrict__ W,
    const float* __restrict__ bias, unsigned short* __restrict__ C,
    int M, int N, int K, int vt)
{
    const int lane = threadIdx.x & 31, wave = threadIdx.x >> 5;
    const int wm = wave >> 2, wn = wave & 3;
    const int m0 = blockIdx.y * 64 + wm * 32;
    const int n0 = blockIdx.x * 128 + wn * 32;
    if (m0 >= M || n0 >= N) return;

    v8f acc[2][2] = {};
    for (int k0 = 0; k0 < K; k0 += 32) {
        v16us a0 = load_a16x32(A + (size_t)m0 * K + k0, K, lane);
        v16us a1 = load_a16x32(A + (size_t)(m0 + 16) * K + k0, K, lane);
        v16us b0 = load_b32x16(W + (size_t)n0 * K + k0, K, lane);
        v16us b1 = load_b32x16(W + (size_t)(n0 + 16) * K + k0, K, lane);
        acc[0][0] = wmma_bf16(a0, b0, acc[0][0]);
        acc[0][1] = wmma_bf16(a0, b1, acc[0][1]);
        acc[1][0] = wmma_bf16(a1, b0, acc[1][0]);
        acc[1][1] = wmma_bf16(a1, b1, acc[1][1]);
    }
    const int n = lane & 15, half = lane >> 4;
    for (int i = 0; i < 2; ++i)
        for (int j = 0; j < 2; ++j) {
            const int gn = n0 + 16 * j + n;
            const float bv = bias[gn];
            for (int v = 0; v < 8; ++v) {
                const int gm = m0 + 16 * i + v + 8 * half;
                const float val = acc[i][j][v] + bv;
                if (!vt) {
                    C[(size_t)gm * N + gn] = f2bf(val);
                } else {
                    const int bI = gm >> 11;           // /TT
                    const int t  = gm & (TT - 1);
                    C[((size_t)bI * DH + gn) * TT + t] = f2bf(val);
                }
            }
        }
}

// ---------------- GEMM fp32 out (final O projection) ----------------
__global__ __launch_bounds__(256) void gemm_f32out_kernel(
    const unsigned short* __restrict__ A, const unsigned short* __restrict__ W,
    const float* __restrict__ bias, float* __restrict__ C,
    int M, int N, int K)
{
    const int lane = threadIdx.x & 31, wave = threadIdx.x >> 5;
    const int wm = wave >> 2, wn = wave & 3;
    const int m0 = blockIdx.y * 64 + wm * 32;
    const int n0 = blockIdx.x * 128 + wn * 32;
    if (m0 >= M || n0 >= N) return;

    v8f acc[2][2] = {};
    for (int k0 = 0; k0 < K; k0 += 32) {
        v16us a0 = load_a16x32(A + (size_t)m0 * K + k0, K, lane);
        v16us a1 = load_a16x32(A + (size_t)(m0 + 16) * K + k0, K, lane);
        v16us b0 = load_b32x16(W + (size_t)n0 * K + k0, K, lane);
        v16us b1 = load_b32x16(W + (size_t)(n0 + 16) * K + k0, K, lane);
        acc[0][0] = wmma_bf16(a0, b0, acc[0][0]);
        acc[0][1] = wmma_bf16(a0, b1, acc[0][1]);
        acc[1][0] = wmma_bf16(a1, b0, acc[1][0]);
        acc[1][1] = wmma_bf16(a1, b1, acc[1][1]);
    }
    const int n = lane & 15, half = lane >> 4;
    for (int i = 0; i < 2; ++i)
        for (int j = 0; j < 2; ++j) {
            const int gn = n0 + 16 * j + n;
            const float bv = bias[gn];
            for (int v = 0; v < 8; ++v) {
                const int gm = m0 + 16 * i + v + 8 * half;
                C[(size_t)gm * N + gn] = acc[i][j][v] + bv;
            }
        }
}

// ---------------- Attention kernel ----------------
// One workgroup per (batch, 16-query tile). 16 waves (512 threads).
// LDS: sQ [16][1032] bf16, sP [16][2056] bf16, sMean [16*2048] f32, sCtx [16][256] f32.
#define SQ_LD   1032
#define SP_LD   2056
#define SQ_SZ   (16 * SQ_LD)                  // ushorts
#define SP_SZ   (16 * SP_LD)                  // ushorts
#define SMEAN_SZ (16 * TT)                    // floats
#define SCTX_SZ (16 * 256)                    // floats
#define ATTN_SMEM_BYTES ((SQ_SZ + SP_SZ) * 2 + (SMEAN_SZ + SCTX_SZ) * 4)

__global__ __launch_bounds__(512) void attn_kernel(
    const unsigned short* __restrict__ Qp,   // [B,T,D] bf16
    const unsigned short* __restrict__ Kp,   // [B,T,D] bf16
    const unsigned short* __restrict__ Vt,   // [B,DH,T] bf16 (transposed V)
    unsigned short* __restrict__ ctx,        // [B,T,D] bf16
    float* __restrict__ out2)                // [B,T,T] f32 head-mean attention
{
    extern __shared__ __align__(16) unsigned char smem[];
    unsigned short* sQ = (unsigned short*)smem;
    unsigned short* sP = sQ + SQ_SZ;
    float* sMean = (float*)(sP + SP_SZ);
    float* sCtx  = sMean + SMEAN_SZ;

    const int b  = blockIdx.x >> 7;          // / (T/16)
    const int q0 = (blockIdx.x & 127) * 16;
    const int tid = threadIdx.x;
    const int wave = tid >> 5, lane = tid & 31;
    const float scale = 0.125f;              // 1/sqrt(64)

    // Load Q tile (16 x 1024 bf16) into LDS (padded rows), zero sMean.
    const unsigned short* Qsrc = Qp + ((size_t)b * TT + q0) * DD;
    for (int i = tid; i < 16 * (DD / 8); i += 512) {     // 8 bf16 per uint4
        const int r = i >> 7, c8 = i & 127;
        *(uint4*)&sQ[r * SQ_LD + c8 * 8] = *(const uint4*)(Qsrc + (size_t)r * DD + c8 * 8);
    }
    for (int i = tid; i < SMEAN_SZ / 4; i += 512)
        ((float4*)sMean)[i] = make_float4(0.f, 0.f, 0.f, 0.f);
    __syncthreads();

    for (int h = 0; h < HH; ++h) {
        const unsigned short* Kh = Kp + (size_t)b * TT * DD + h * DH;

        // ---- Phase A: S = scale * Q_h K_h^T, wave w covers 8 k-tiles ----
        for (int i = 0; i < 8; ++i) {
            const int t0 = (wave * 8 + i) * 16;
            v8f acc = {};
            for (int kk = 0; kk < DH; kk += 32) {
                v16us a = load_a16x32(sQ + h * DH + kk, SQ_LD, lane);
                v16us bb = load_b32x16(Kh + (size_t)t0 * DD + kk, DD, lane);
                acc = wmma_bf16(a, bb, acc);
            }
            const int n = lane & 15, half = lane >> 4;
            for (int v = 0; v < 8; ++v) {
                const int m = v + 8 * half;
                sP[m * SP_LD + t0 + n] = f2bf(acc[v] * scale);
            }
        }
        __syncthreads();

        // ---- Softmax: wave r handles row r (shuffle reductions) ----
        {
            const int r = wave;
            float mx = -3.0e38f;
            for (int c = lane; c < TT; c += 32) mx = fmaxf(mx, bf2f(sP[r * SP_LD + c]));
            for (int off = 16; off; off >>= 1) mx = fmaxf(mx, __shfl_xor(mx, off, 32));
            float sum = 0.f;
            for (int c = lane; c < TT; c += 32) {
                const float e = __expf(bf2f(sP[r * SP_LD + c]) - mx);
                sum += e;
                sP[r * SP_LD + c] = f2bf(e);
            }
            for (int off = 16; off; off >>= 1) sum += __shfl_xor(sum, off, 32);
            const float inv = 1.0f / sum;
            const float invh = inv * (1.0f / (float)HH);
            for (int c = lane; c < TT; c += 32) {
                const float e = bf2f(sP[r * SP_LD + c]);
                sP[r * SP_LD + c] = f2bf(e * inv);
                sMean[r * TT + c] += e * invh;
            }
        }
        __syncthreads();

        // ---- Phase B: ctx_h = P @ Vs (wave -> n-tile x k-quarter) ----
        {
            const int nt = wave & 3, kq = wave >> 2;
            const unsigned short* Vh = Vt + ((size_t)b * DH + nt * 16) * TT;
            v8f acc = {};
            for (int s = 0; s < 16; ++s) {
                const int k0 = kq * 512 + s * 32;
                v16us a = load_a16x32(sP + k0, SP_LD, lane);
                v16us bb = load_b32x16(Vh + k0, TT, lane);
                acc = wmma_bf16(a, bb, acc);
            }
            const int n = lane & 15, half = lane >> 4;
            for (int v = 0; v < 8; ++v) {
                const int m = v + 8 * half;
                sCtx[wave * 256 + m * 16 + n] = acc[v];
            }
        }
        __syncthreads();

        // ---- Reduce k-quarters, store ctx tile ----
        for (int i = tid; i < 1024; i += 512) {
            const int nt = i >> 8;
            const int mn = i & 255;
            const int m = mn >> 4, n = mn & 15;
            const float s = sCtx[(0 * 4 + nt) * 256 + mn] + sCtx[(1 * 4 + nt) * 256 + mn] +
                            sCtx[(2 * 4 + nt) * 256 + mn] + sCtx[(3 * 4 + nt) * 256 + mn];
            ctx[((size_t)b * TT + q0 + m) * DD + h * DH + nt * 16 + n] = f2bf(s);
        }
        __syncthreads();
    }

    // ---- Store head-averaged attention (contiguous 16 rows) ----
    float* o2 = out2 + ((size_t)b * TT + q0) * TT;
    for (int i = tid; i < SMEAN_SZ / 4; i += 512)
        ((float4*)o2)[i] = ((const float4*)sMean)[i];
}

// ---------------- Host launch ----------------
extern "C" void kernel_launch(void* const* d_in, const int* in_sizes, int n_in,
                              void* d_out, int out_size, void* d_ws, size_t ws_size,
                              hipStream_t stream) {
    const float* q  = (const float*)d_in[0];
    const float* k  = (const float*)d_in[1];
    const float* v  = (const float*)d_in[2];
    const float* Wq = (const float*)d_in[3];
    const float* bq = (const float*)d_in[4];
    const float* Wk = (const float*)d_in[5];
    const float* bk = (const float*)d_in[6];
    const float* Wv = (const float*)d_in[7];
    const float* bv = (const float*)d_in[8];
    const float* Wo = (const float*)d_in[9];
    const float* bo = (const float*)d_in[10];

    const size_t NTOK = (size_t)BB * TT;          // 8192
    const size_t NQKV = NTOK * DD;                // 8,388,608
    const size_t NW   = (size_t)DD * DD;          // 1,048,576
    const size_t NWV  = (size_t)DH * DD;          // 65,536

    unsigned char* ws = (unsigned char*)d_ws;
    size_t off = 0;
    auto carve = [&](size_t bytes) -> unsigned char* {
        unsigned char* p = ws + off;
        off += (bytes + 255) & ~(size_t)255;
        return p;
    };
    unsigned short* qb  = (unsigned short*)carve(NQKV * 2);
    unsigned short* kb  = (unsigned short*)carve(NQKV * 2);
    unsigned short* vb  = (unsigned short*)carve(NQKV * 2);
    unsigned short* Wqb = (unsigned short*)carve(NW * 2);
    unsigned short* Wkb = (unsigned short*)carve(NW * 2);
    unsigned short* Wvb = (unsigned short*)carve(NWV * 2);
    unsigned short* Wob = (unsigned short*)carve(NW * 2);
    unsigned short* Qp  = (unsigned short*)carve(NQKV * 2);
    unsigned short* Kpp = (unsigned short*)carve(NQKV * 2);
    unsigned short* Vt  = (unsigned short*)carve((size_t)BB * DH * TT * 2);
    unsigned short* ctx = (unsigned short*)carve(NQKV * 2);

    float* y    = (float*)d_out;                  // [B,T,D]
    float* out2 = y + NQKV;                       // [B,T,T]

    // 1) fp32 -> bf16 converts
    cvt_bf16_kernel<<<2048, 256, 0, stream>>>(q,  qb,  (int)NQKV);
    cvt_bf16_kernel<<<2048, 256, 0, stream>>>(k,  kb,  (int)NQKV);
    cvt_bf16_kernel<<<2048, 256, 0, stream>>>(v,  vb,  (int)NQKV);
    cvt_bf16_kernel<<<1024, 256, 0, stream>>>(Wq, Wqb, (int)NW);
    cvt_bf16_kernel<<<1024, 256, 0, stream>>>(Wk, Wkb, (int)NW);
    cvt_bf16_kernel<<<256,  256, 0, stream>>>(Wv, Wvb, (int)NWV);
    cvt_bf16_kernel<<<1024, 256, 0, stream>>>(Wo, Wob, (int)NW);

    // 2) Projections (WMMA bf16 GEMMs)
    dim3 gqk(DD / 128, (unsigned)(NTOK / 64));    // (8,128)
    gemm_bf16out_kernel<<<gqk, 256, 0, stream>>>(qb, Wqb, bq, Qp,  (int)NTOK, DD, DD, 0);
    gemm_bf16out_kernel<<<gqk, 256, 0, stream>>>(kb, Wkb, bk, Kpp, (int)NTOK, DD, DD, 0);
    dim3 gv(1, (unsigned)(NTOK / 64));            // (1,128), N=64 -> half waves idle
    gemm_bf16out_kernel<<<gv, 256, 0, stream>>>(vb, Wvb, bv, Vt, (int)NTOK, DH, DD, 1);

    // 3) Attention (WMMA QK^T + softmax + WMMA PV + head-mean)
    attn_kernel<<<BB * (TT / 16), 512, ATTN_SMEM_BYTES, stream>>>(Qp, Kpp, Vt, ctx, out2);

    // 4) Output projection (WMMA, fp32 out)
    gemm_f32out_kernel<<<gqk, 256, 0, stream>>>(ctx, Wob, bo, y, (int)NTOK, DD, DD);
}
